// Attention_86981677679321
// MI455X (gfx1250) — compile-verified, matrix operand order checked by reading, not run on hardware
//
#include <hip/hip_runtime.h>
#include <hip/hip_bf16.h>

// ---------------------------------------------------------------------------
// CDNA5 (gfx1250) attention block:  out = softmax(QK^T * s) V  projected.
// All GEMM-like math goes through v_wmma_f32_16x16x32_f16 (f16 in, f32 acc).
// V is pre-transposed to [B][C][N] so every WMMA fragment load is two
// contiguous global_load_b128s; next-block K/V rows are prefetched with
// global_prefetch_b8. Fragment loads are hoisted into distinct registers so
// WMMAs overlap outstanding loads instead of waiting on loadcnt==0.
// ---------------------------------------------------------------------------

typedef __attribute__((ext_vector_type(16))) _Float16 v16h;
typedef __attribute__((ext_vector_type(8)))  _Float16 v8h;
typedef __attribute__((ext_vector_type(8)))  float    v8f;

// Problem constants (fixed by the reference)
#define BB   4
#define NN   2048
#define CC   1024
#define HH   16
#define DD   64
#define LD3  (3 * CC)   // 3072, row stride of qkv buffer
#define MM   (BB * NN)  // 8192 tokens

__device__ __forceinline__ v8f wmma32(v16h a, v16h b, v8f c) {
    // (neg_a, A, neg_b, B, c_mod, C, reuse_a, reuse_b)
    return __builtin_amdgcn_wmma_f32_16x16x32_f16(
        false, a, false, b, (short)0, c, false, false);
}

// Load a 16x32 f16 fragment in the CDNA5 16-bit A/B layout from a row-major
// matrix (16-wide index across lanes&15; K contiguous within a row).
//   lanes 0-15 : idx=lane,    halves [0..7]=K0..7,  [8..15]=K16..23
//   lanes 16-31: idx=lane-16, halves [0..7]=K8..15, [8..15]=K24..31
__device__ __forceinline__ v16h load_frag_rowmajor(const _Float16* base, int ld) {
    const int lane = threadIdx.x & 31;
    const _Float16* p = base + (size_t)(lane & 15) * ld + ((lane >> 4) << 3);
    v8h lo = *(const v8h*)(p);
    v8h hi = *(const v8h*)(p + 16);
    return __builtin_shufflevector(lo, hi, 0, 1, 2, 3, 4, 5, 6, 7,
                                           8, 9, 10, 11, 12, 13, 14, 15);
}

// ---------------------------------------------------------------------------
// Elementwise f32 -> f16 cast
// ---------------------------------------------------------------------------
__global__ void cast_f16_kernel(const float* __restrict__ in,
                                _Float16* __restrict__ out, int n) {
    int i = blockIdx.x * blockDim.x + threadIdx.x;
    if (i < n) out[i] = (_Float16)in[i];
}

// Transposing cast: in[rows][cols] (f32) -> out[cols][rows] (f16).
__global__ void transpose_cast_f16_kernel(const float* __restrict__ in,
                                          _Float16* __restrict__ out,
                                          int rows, int cols) {
    int i = blockIdx.x * blockDim.x + threadIdx.x;
    if (i < rows * cols) {
        int r = i / cols;
        int c = i - r * cols;
        out[(size_t)c * rows + r] = (_Float16)in[i];
    }
}

// Extract + transpose the V slab of qkv:
//   vT[b*C + c][n] = qkv[(b*N + n)*3C + 2C + c]
__global__ void transpose_v_kernel(const _Float16* __restrict__ qkv,
                                   _Float16* __restrict__ vT) {
    int i = blockIdx.x * blockDim.x + threadIdx.x;  // over MM*CC
    if (i < MM * CC) {
        const int tok = i / CC;          // b*N + n
        const int c   = i - tok * CC;    // h*64 + d
        const int b   = tok / NN;
        const int n   = tok - b * NN;
        vT[((size_t)b * CC + c) * NN + n] =
            qkv[(size_t)tok * LD3 + 2 * CC + c];
    }
}

// ---------------------------------------------------------------------------
// WMMA GEMM:  C[M,Nn] = A[M,K] * B[K,Nn]   (B supplied transposed: BT[Nn][K])
// One wave computes a 16x64 tile (4 accumulators), K-loop step 32.
// ---------------------------------------------------------------------------
template <bool OUT_F16>
__global__ __launch_bounds__(32)
void wmma_gemm_kernel(const _Float16* __restrict__ A,
                      const _Float16* __restrict__ BT,
                      _Float16* __restrict__ Ch,
                      float* __restrict__ Cf,
                      const float* __restrict__ bias,
                      int M, int K, int Nn) {
    const int n0 = blockIdx.x * 64;
    const int m0 = blockIdx.y * 16;

    const _Float16* Abase = A + (size_t)m0 * K;

    const v8f vzero = {0.f, 0.f, 0.f, 0.f, 0.f, 0.f, 0.f, 0.f};
    v8f acc[4];
#pragma unroll
    for (int j = 0; j < 4; ++j) acc[j] = vzero;

    const int lane = threadIdx.x & 31;
    const size_t lrow = (size_t)(lane & 15) * K + ((lane >> 4) << 3);

#pragma unroll 2
    for (int k0 = 0; k0 < K; k0 += 32) {
        // prefetch next K-step's A rows (one line per lane)
        if (k0 + 32 < K)
            __builtin_prefetch(Abase + lrow + k0 + 32, 0, 0);
        v16h a  = load_frag_rowmajor(Abase + k0, K);
        v16h b0 = load_frag_rowmajor(BT + (size_t)(n0 +  0) * K + k0, K);
        v16h b1 = load_frag_rowmajor(BT + (size_t)(n0 + 16) * K + k0, K);
        v16h b2 = load_frag_rowmajor(BT + (size_t)(n0 + 32) * K + k0, K);
        v16h b3 = load_frag_rowmajor(BT + (size_t)(n0 + 48) * K + k0, K);
        acc[0] = wmma32(a, b0, acc[0]);
        acc[1] = wmma32(a, b1, acc[1]);
        acc[2] = wmma32(a, b2, acc[2]);
        acc[3] = wmma32(a, b3, acc[3]);
    }

    // C/D layout: reg r, lane -> (row = r + 8*(lane>>4), col = lane&15)
    const int half = lane >> 4;
    const int col  = lane & 15;
#pragma unroll
    for (int j = 0; j < 4; ++j) {
#pragma unroll
        for (int r = 0; r < 8; ++r) {
            const int row = m0 + r + 8 * half;
            const int cn  = n0 + j * 16 + col;
            if (OUT_F16) {
                Ch[(size_t)row * Nn + cn] = (_Float16)acc[j][r];
            } else {
                Cf[(size_t)row * Nn + cn] = acc[j][r] + bias[cn];
            }
        }
    }
}

// ---------------------------------------------------------------------------
// Flash attention: one wave per (batch, head, 16-query tile).
// qkv layout: [B*N, 3C] with q @ col h*64, k @ col C + h*64.
// V comes from the transposed slab vT[B*C][N].
// ---------------------------------------------------------------------------
__global__ __launch_bounds__(32)
void attention_kernel(const _Float16* __restrict__ qkv,
                      const _Float16* __restrict__ vT,
                      _Float16* __restrict__ outh) {
    const int q0   = blockIdx.x * 16;
    const int h    = blockIdx.y;
    const int b    = blockIdx.z;
    const size_t tokBase = (size_t)b * NN;

    const int lane = threadIdx.x & 31;
    const int half = lane >> 4;
    const int col  = lane & 15;

    __shared__ _Float16 Pld[16 * 32];   // prob tile staging: C-layout -> A-layout

    // Q fragments (16 rows x 64 d), two K=32 chunks
    const _Float16* qbase = qkv + (tokBase + q0) * LD3 + h * DD;
    const v16h qa0 = load_frag_rowmajor(qbase, LD3);
    const v16h qa1 = load_frag_rowmajor(qbase + 32, LD3);

    const _Float16* kbase0 = qkv + tokBase * LD3 + CC + h * DD;   // keys, token-major
    const _Float16* vtbase = vT + ((size_t)b * CC + h * DD) * NN; // [64][NN]

    const v8f vzero = {0.f, 0.f, 0.f, 0.f, 0.f, 0.f, 0.f, 0.f};
    v8f o[4];
#pragma unroll
    for (int j = 0; j < 4; ++j) o[j] = vzero;

    float m[8], l[8];
#pragma unroll
    for (int r = 0; r < 8; ++r) { m[r] = -1e30f; l[r] = 0.f; }

    const float scale = 0.125f;  // D^-0.5 = 64^-0.5

    for (int kb = 0; kb < NN; kb += 32) {
        const _Float16* kbase = kbase0 + (size_t)kb * LD3;

        // prefetch next key block's K rows and vT rows (one line per lane)
        if (kb + 32 < NN) {
            __builtin_prefetch(
                kbase + (size_t)(32 + (lane & 15)) * LD3 + ((lane >> 4) << 3), 0, 0);
            __builtin_prefetch(
                vtbase + (size_t)(lane & 15) * NN + kb + 32, 0, 0);
        }

        // ---- issue ALL loads for this block into distinct registers ----
        v16h kf00 = load_frag_rowmajor(kbase, LD3);                    // keys +0,  d 0..31
        v16h kf01 = load_frag_rowmajor(kbase + 32, LD3);               // keys +0,  d 32..63
        v16h kf10 = load_frag_rowmajor(kbase + (size_t)16 * LD3, LD3); // keys +16, d 0..31
        v16h kf11 = load_frag_rowmajor(kbase + (size_t)16 * LD3 + 32, LD3);
        // V fragments are independent of the scores: issue now, consume after
        // softmax so the ds_bpermute/exp work hides their latency.
        v16h vf0 = load_frag_rowmajor(vtbase + (size_t) 0 * NN + kb, NN);
        v16h vf1 = load_frag_rowmajor(vtbase + (size_t)16 * NN + kb, NN);
        v16h vf2 = load_frag_rowmajor(vtbase + (size_t)32 * NN + kb, NN);
        v16h vf3 = load_frag_rowmajor(vtbase + (size_t)48 * NN + kb, NN);

        // ---- scores: S[16 x 32] as two 16x16 f32 tiles ----
        v8f s0 = vzero, s1 = vzero;
        s0 = wmma32(qa0, kf00, s0);
        s0 = wmma32(qa1, kf01, s0);
        s1 = wmma32(qa0, kf10, s1);
        s1 = wmma32(qa1, kf11, s1);

        // ---- online softmax over this 32-key block ----
        float fac[8];
#pragma unroll
        for (int r = 0; r < 8; ++r) {
            float a0 = s0[r] * scale;
            float a1 = s1[r] * scale;
            float rm = fmaxf(a0, a1);
            rm = fmaxf(rm, __shfl_xor(rm, 1, 32));
            rm = fmaxf(rm, __shfl_xor(rm, 2, 32));
            rm = fmaxf(rm, __shfl_xor(rm, 4, 32));
            rm = fmaxf(rm, __shfl_xor(rm, 8, 32));   // row max within 16-lane half
            const float nm = fmaxf(m[r], rm);
            const float p0 = __expf(a0 - nm);
            const float p1 = __expf(a1 - nm);
            float rs = p0 + p1;
            rs += __shfl_xor(rs, 1, 32);
            rs += __shfl_xor(rs, 2, 32);
            rs += __shfl_xor(rs, 4, 32);
            rs += __shfl_xor(rs, 8, 32);             // row sum within half
            const float f = __expf(m[r] - nm);
            l[r] = l[r] * f + rs;
            m[r] = nm;
            fac[r] = f;
            // stash probabilities (C/D layout -> LDS row-major 16x32)
            const int row = r + 8 * half;
            Pld[row * 32 + col]      = (_Float16)p0;
            Pld[row * 32 + 16 + col] = (_Float16)p1;
        }

        // rescale running output accumulators
#pragma unroll
        for (int j = 0; j < 4; ++j)
#pragma unroll
            for (int r = 0; r < 8; ++r) o[j][r] *= fac[r];

        __syncthreads();  // make LDS prob tile visible / ordered
        const v16h pa = load_frag_rowmajor(&Pld[0], 32);

        // ---- O += P[16x32] * V[32x64] ----
        o[0] = wmma32(pa, vf0, o[0]);
        o[1] = wmma32(pa, vf1, o[1]);
        o[2] = wmma32(pa, vf2, o[2]);
        o[3] = wmma32(pa, vf3, o[3]);
        __syncthreads();
    }

    // ---- epilogue: normalize and store f16 at [token, h*64 + d] ----
    _Float16* obase = outh + (tokBase + q0) * CC + h * DD;
#pragma unroll
    for (int j = 0; j < 4; ++j) {
#pragma unroll
        for (int r = 0; r < 8; ++r) {
            const int row = r + 8 * half;
            obase[(size_t)row * CC + j * 16 + col] = (_Float16)(o[j][r] / l[r]);
        }
    }
}

// ---------------------------------------------------------------------------
// Host-side launch
// ---------------------------------------------------------------------------
extern "C" void kernel_launch(void* const* d_in, const int* in_sizes, int n_in,
                              void* d_out, int out_size, void* d_ws, size_t ws_size,
                              hipStream_t stream) {
    const float* x      = (const float*)d_in[0];   // [B,N,C]
    const float* W_qkv  = (const float*)d_in[1];   // [C, 3C]
    const float* W_proj = (const float*)d_in[2];   // [C, C]
    const float* b_proj = (const float*)d_in[3];   // [C]
    float* out = (float*)d_out;                    // [B,N,C] f32

    // Workspace layout (all f16)
    _Float16* xh     = (_Float16*)d_ws;                 // [M, C]
    _Float16* wqkvT  = xh     + (size_t)MM * CC;        // [3C, C]
    _Float16* wprojT = wqkvT  + (size_t)CC * 3 * CC;    // [C, C]
    _Float16* qkvh   = wprojT + (size_t)CC * CC;        // [M, 3C]
    _Float16* attnh  = qkvh   + (size_t)MM * 3 * CC;    // [M, C]
    _Float16* vT     = attnh  + (size_t)MM * CC;        // [B*C, N]

    // 1) precision casts (weights transposed for contiguous B-fragments)
    {
        const int nx = MM * CC;
        cast_f16_kernel<<<(nx + 255) / 256, 256, 0, stream>>>(x, xh, nx);
        const int nq = CC * 3 * CC;
        transpose_cast_f16_kernel<<<(nq + 255) / 256, 256, 0, stream>>>(
            W_qkv, wqkvT, CC, 3 * CC);
        const int np = CC * CC;
        transpose_cast_f16_kernel<<<(np + 255) / 256, 256, 0, stream>>>(
            W_proj, wprojT, CC, CC);
    }

    // 2) qkv = x @ W_qkv              (f16 out)
    wmma_gemm_kernel<true><<<dim3(3 * CC / 64, MM / 16), 32, 0, stream>>>(
        xh, wqkvT, qkvh, nullptr, nullptr, MM, CC, 3 * CC);

    // 2b) transpose V slab for contiguous P*V fragment loads
    {
        const int nv = MM * CC;
        transpose_v_kernel<<<(nv + 255) / 256, 256, 0, stream>>>(qkvh, vT);
    }

    // 3) flash attention per (b, h, 16-query tile)
    attention_kernel<<<dim3(NN / 16, HH, BB), 32, 0, stream>>>(qkvh, vT, attnh);

    // 4) out = attn @ W_proj + b_proj (f32 out)
    wmma_gemm_kernel<false><<<dim3(CC / 64, MM / 16), 32, 0, stream>>>(
        attnh, wprojT, nullptr, out, b_proj, MM, CC, CC);
}
